// LSTMTagger_3401614098573
// MI455X (gfx1250) — compile-verified, matrix operand order checked by reading, not run on hardware
//
#include <hip/hip_runtime.h>
#include <hip/hip_bf16.h>
#include <math.h>

// ---------------------------------------------------------------------------
// MI455X (gfx1250) LSTM tagger.
//   S=4096, C=12, V=50000, NC=26, E=1024, EC=256, H=1024, HC=256, T=64
// GEMMs: v_wmma_f32_16x16x32_bf16, 32x64 wave tile, register double-buffered.
// Word recurrence: persistent-RNN style — W_hh lives in VGPRs for the whole
// kernel (each lane owns a quarter row, 128 packed uints), h broadcast from
// LDS, one atomic device barrier (+ s_cluster_barrier fast path) per step.
// ---------------------------------------------------------------------------

typedef __attribute__((ext_vector_type(16))) __bf16 v16bf;
typedef __attribute__((ext_vector_type(8)))  __bf16 v8bf;
typedef __attribute__((ext_vector_type(8)))  float  v8f;

#define S_LEN 4096
#define C_LEN 12
#define EC_D  256
#define HC_D  256
#define E_D   1024
#define H_D   1024
#define T_D   64
#define NWG_SEQ 64          // 64 WGs x 64 gate rows = 4096 gates

__device__ __forceinline__ unsigned short f2bf(float f) {
  unsigned u = __float_as_uint(f);
  u += 0x7FFFu + ((u >> 16) & 1u);          // round-to-nearest-even
  return (unsigned short)(u >> 16);
}
__device__ __forceinline__ float bf2f(unsigned short b) {
  return __uint_as_float(((unsigned)b) << 16);
}
__device__ __forceinline__ float sigmoidf(float x) { return 1.0f / (1.0f + __expf(-x)); }

// ---------------------------------------------------------------- weight prep
__global__ void cvt_bf16_strided(const float* __restrict__ src, int srcLd,
                                 unsigned short* __restrict__ dst, int dstLd,
                                 int rows, int cols) {
  long total = (long)rows * cols;
  for (long i = (long)blockIdx.x * blockDim.x + threadIdx.x; i < total;
       i += (long)gridDim.x * blockDim.x) {
    int r = (int)(i / cols), c = (int)(i % cols);
    dst[(long)r * dstLd + c] = f2bf(src[(long)r * srcLd + c]);
  }
}

__global__ void add_bias_vec(const float* __restrict__ a, const float* __restrict__ b,
                             float* __restrict__ out, int n) {
  for (int i = blockIdx.x * blockDim.x + threadIdx.x; i < n; i += gridDim.x * blockDim.x)
    out[i] = a[i] + b[i];
}

// ------------------------------------------------------------- embeddings
// A_char: 12 buffers of [4096 x 512] bf16; cols 0:256 = char embedding at step t,
// cols 256:512 = recurrent h (written by cell kernel; step 0 zeroed here).
__global__ void embed_chars(const int* __restrict__ char_idx,
                            const float* __restrict__ Wec,
                            unsigned short* __restrict__ Achar) {
  const long total1 = (long)S_LEN * C_LEN * EC_D;
  const long total2 = (long)S_LEN * HC_D;
  for (long i = (long)blockIdx.x * blockDim.x + threadIdx.x; i < total1 + total2;
       i += (long)gridDim.x * blockDim.x) {
    if (i < total1) {
      int e = (int)(i & (EC_D - 1));
      int t = (int)((i >> 8) % C_LEN);
      int s = (int)(i / (256 * C_LEN));
      int w = char_idx[s * C_LEN + t];
      Achar[(long)t * S_LEN * 512 + (long)s * 512 + e] = f2bf(Wec[(long)w * EC_D + e]);
    } else {
      long j = i - total1;
      int e = (int)(j & (HC_D - 1));
      int s = (int)(j >> 8);
      Achar[(long)s * 512 + 256 + e] = 0;   // h0 = 0 (bf16 zero)
    }
  }
}

// A_w: [4096 x 1280] bf16; cols 0:1024 = word embedding, cols 1024:1280 = h_char.
__global__ void embed_words(const int* __restrict__ word_idx,
                            const float* __restrict__ Wew,
                            unsigned short* __restrict__ Aw) {
  const long total = (long)S_LEN * E_D;
  for (long i = (long)blockIdx.x * blockDim.x + threadIdx.x; i < total;
       i += (long)gridDim.x * blockDim.x) {
    int e = (int)(i & (E_D - 1));
    int s = (int)(i >> 10);
    Aw[(long)s * 1280 + e] = f2bf(Wew[(long)word_idx[s] * E_D + e]);
  }
}

// ------------------------------------------------------------- WMMA GEMM
// D[M,N] = A[M,K](bf16) * B[N,K](bf16)^T, f32 out; + bias[N] only on rows 0-15 of each 32-row wave tile.
// Block = 128 threads (4 waves) all stacked in M: block tile 128M x 64N,
// wave tile 32M x 64N = 8 WMMA accumulators, register double-buffered.
__device__ __forceinline__ v16bf load_frag(const unsigned short* p) {
  v8bf x0 = *(const v8bf*)(p);
  v8bf x1 = *(const v8bf*)(p + 16);
  v16bf f;
#pragma unroll
  for (int i = 0; i < 8; ++i) { f[i] = x0[i]; f[i + 8] = x1[i]; }
  return f;
}

#define WMMA_BF16(A_, B_, C_) \
  __builtin_amdgcn_wmma_f32_16x16x32_bf16(false, (A_), false, (B_), (short)0, (C_), false, false)

__global__ __launch_bounds__(128) void gemm_bf16_wmma(
    const unsigned short* __restrict__ A, int lda,
    const unsigned short* __restrict__ B, int ldb,
    const float* __restrict__ bias,
    float* __restrict__ D, int ldc, int K) {
  const int lane = threadIdx.x & 31;
  const int wave = threadIdx.x >> 5;
  const int m0 = blockIdx.x * 128 + wave * 32;
  const int n0 = blockIdx.y * 64;
  const int row = lane & 15;
  const int kh  = (lane >> 4) << 3;         // 0 or 8

  const unsigned short* ap0 = A + (long)(m0 + row) * lda + kh;
  const unsigned short* ap1 = ap0 + (long)16 * lda;
  const unsigned short* bp0 = B + (long)(n0 + row) * ldb + kh;
  const unsigned short* bp1 = bp0 + (long)16 * ldb;
  const unsigned short* bp2 = bp0 + (long)32 * ldb;
  const unsigned short* bp3 = bp0 + (long)48 * ldb;

  v8f acc[2][4];
#pragma unroll
  for (int jj = 0; jj < 4; ++jj) {
    float bv = bias ? bias[n0 + jj * 16 + row] : 0.0f;
#pragma unroll
    for (int v = 0; v < 8; ++v) { acc[0][jj][v] = bv; acc[1][jj][v] = 0.0f; }
  }

  v16bf a0 = load_frag(ap0), a1 = load_frag(ap1);
  v16bf b0 = load_frag(bp0), b1 = load_frag(bp1);
  v16bf b2 = load_frag(bp2), b3 = load_frag(bp3);

  for (int k0 = 32; k0 < K; k0 += 32) {
    v16bf na0 = load_frag(ap0 + k0);
    v16bf na1 = load_frag(ap1 + k0);
    v16bf nb0 = load_frag(bp0 + k0);
    v16bf nb1 = load_frag(bp1 + k0);
    v16bf nb2 = load_frag(bp2 + k0);
    v16bf nb3 = load_frag(bp3 + k0);
    if (k0 + 32 < K) {
      __builtin_prefetch(ap0 + k0 + 32, 0, 1);      // global_prefetch_b8
      __builtin_prefetch(bp0 + k0 + 32, 0, 1);
    }
    acc[0][0] = WMMA_BF16(a0, b0, acc[0][0]);
    acc[0][1] = WMMA_BF16(a0, b1, acc[0][1]);
    acc[0][2] = WMMA_BF16(a0, b2, acc[0][2]);
    acc[0][3] = WMMA_BF16(a0, b3, acc[0][3]);
    acc[1][0] = WMMA_BF16(a1, b0, acc[1][0]);
    acc[1][1] = WMMA_BF16(a1, b1, acc[1][1]);
    acc[1][2] = WMMA_BF16(a1, b2, acc[1][2]);
    acc[1][3] = WMMA_BF16(a1, b3, acc[1][3]);
    a0 = na0; a1 = na1; b0 = nb0; b1 = nb1; b2 = nb2; b3 = nb3;
  }
  acc[0][0] = WMMA_BF16(a0, b0, acc[0][0]);
  acc[0][1] = WMMA_BF16(a0, b1, acc[0][1]);
  acc[0][2] = WMMA_BF16(a0, b2, acc[0][2]);
  acc[0][3] = WMMA_BF16(a0, b3, acc[0][3]);
  acc[1][0] = WMMA_BF16(a1, b0, acc[1][0]);
  acc[1][1] = WMMA_BF16(a1, b1, acc[1][1]);
  acc[1][2] = WMMA_BF16(a1, b2, acc[1][2]);
  acc[1][3] = WMMA_BF16(a1, b3, acc[1][3]);

  const int voff = (lane >> 4) << 3;
#pragma unroll
  for (int mi = 0; mi < 2; ++mi) {
    const int rbase = m0 + mi * 16 + voff;
#pragma unroll
    for (int jj = 0; jj < 4; ++jj) {
      int col = n0 + jj * 16 + row;
#pragma unroll
      for (int v = 0; v < 8; ++v)
        D[(long)(rbase + v) * ldc + col] = acc[mi][jj][v];
    }
  }
}

// ------------------------------------------------------------- char LSTM cell
__global__ void lstm_cell_char(const float* __restrict__ G,
                               float* __restrict__ Cst,
                               unsigned short* __restrict__ Hdst, int ldH,
                               int first) {
  const long total = (long)S_LEN * HC_D;
  for (long i = (long)blockIdx.x * blockDim.x + threadIdx.x; i < total;
       i += (long)gridDim.x * blockDim.x) {
    int j = (int)(i & (HC_D - 1));
    int s = (int)(i >> 8);
    const float* g = G + (long)s * 1024;
    float ig = sigmoidf(g[j]);
    float fg = sigmoidf(g[256 + j]);
    float gg = tanhf(g[512 + j]);
    float og = sigmoidf(g[768 + j]);
    float cp = first ? 0.0f : Cst[i];
    float c = fg * cp + ig * gg;
    float h = og * tanhf(c);
    Cst[i] = c;
    Hdst[(long)s * ldH + j] = f2bf(h);
  }
}

// ------------------------------------------------------------- barrier init
__global__ void init_barrier(unsigned* bar) {
  if (blockIdx.x == 0 && threadIdx.x == 0) { bar[0] = 0u; bar[1] = 0u; }
}

// ------------------------------------------------------------- word LSTM (seq)
__device__ __forceinline__ void grid_barrier(unsigned* bar) {
  __syncthreads();
  __threadfence();                                   // device-scope release
  if (threadIdx.x == 0) {
    // Fast-path if dispatched as a workgroup cluster (NOP when ClusterID==0):
    __builtin_amdgcn_s_cluster_barrier();            // s_barrier_signal/-wait -3
    unsigned gen = __hip_atomic_load(&bar[1], __ATOMIC_ACQUIRE, __HIP_MEMORY_SCOPE_AGENT);
    unsigned arr = __hip_atomic_fetch_add(&bar[0], 1u, __ATOMIC_ACQ_REL, __HIP_MEMORY_SCOPE_AGENT);
    if (arr == NWG_SEQ - 1) {
      __hip_atomic_store(&bar[0], 0u, __ATOMIC_RELAXED, __HIP_MEMORY_SCOPE_AGENT);
      __hip_atomic_fetch_add(&bar[1], 1u, __ATOMIC_ACQ_REL, __HIP_MEMORY_SCOPE_AGENT);
    } else {
      while (__hip_atomic_load(&bar[1], __ATOMIC_ACQUIRE, __HIP_MEMORY_SCOPE_AGENT) == gen)
        __builtin_amdgcn_s_sleep(1);
    }
  }
  __syncthreads();
}

// Persistent-RNN word recurrence. 64 WGs x 256 threads.
// WG w owns gate rows [w*64, w*64+64). Each lane owns a QUARTER of one W_hh
// row: 256 bf16 = 128 packed uints, held in VGPRs for the whole kernel
// (fully-unrolled local array -> SROA; ~170 VGPRs/lane, fits well under the
// 256 directly-addressable VGPRs). Per step the matvec is register x LDS:
// lanes of the same quarter read identical LDS addresses (broadcast), then a
// 2-hop shfl_xor combines the 4 quarters. W_hh is read from memory ONCE.
__global__ __launch_bounds__(256, 2) void word_lstm_seq(
    const float* __restrict__ P,               // [4096 x 4096] x@W_ih^T, + b only where row % 32 < 16
    const unsigned short* __restrict__ Whh,    // [4096 x 1024] bf16
    float* __restrict__ gates_buf,             // [2 x 4096] ping-pong
    unsigned short* __restrict__ hs,           // [4096 x 1024] bf16
    unsigned* __restrict__ bar) {
  __shared__ float hsm[H_D];
  __shared__ float cfull[H_D];
  const int wg   = blockIdx.x;
  const int tid  = threadIdx.x;
  const int lane = tid & 31;
  const int wave = tid >> 5;
  const int q    = lane & 3;                   // quarter of the row
  const int n    = wg * 64 + wave * 8 + (lane >> 2);   // owned gate row

  // ---- pin this lane's quarter-row of W_hh in registers (one-time load)
  const unsigned* wsrc = (const unsigned*)Whh + (long)n * 512 + q * 128;
  unsigned wreg[128];
#pragma unroll
  for (int i = 0; i < 128; ++i) wreg[i] = wsrc[i];

  for (int i = tid; i < H_D; i += 256) { hsm[i] = 0.0f; cfull[i] = 0.0f; }
  __syncthreads();

  const float2* hp2 = (const float2*)hsm + q * 128;   // this quarter's h pairs

  for (int t = 0; t < S_LEN; ++t) {
    float* gb = gates_buf + ((t & 1) << 12);
    // ---- phase A: gates[n] = P[t][n] + dot(h, W_hh[n])  (weights in VGPRs)
    float s0 = 0.0f, s1 = 0.0f;
#pragma unroll
    for (int i = 0; i < 128; ++i) {
      unsigned w = wreg[i];
      float2 hv = hp2[i];                      // ds_load_b64, broadcast
      s0 += bf2f((unsigned short)(w & 0xFFFFu)) * hv.x;
      s1 += bf2f((unsigned short)(w >> 16)) * hv.y;
    }
    float sum = s0 + s1;
    sum += __shfl_xor(sum, 1, 32);             // combine 4 quarters
    sum += __shfl_xor(sum, 2, 32);
    if (q == 0) gb[n] = sum + P[(long)t * 4096 + n];
    grid_barrier(bar);                         // single barrier per step

    // ---- cell update, replicated on every WG (j covers all 1024 hidden)
    for (int j = tid; j < H_D; j += 256) {
      float ig = sigmoidf(gb[j]);
      float fg = sigmoidf(gb[1024 + j]);
      float gg = tanhf(gb[2048 + j]);
      float og = sigmoidf(gb[3072 + j]);
      float c = fg * cfull[j] + ig * gg;
      cfull[j] = c;
      float h = og * tanhf(c);
      hsm[j] = h;
      if ((j >> 4) == wg) hs[(long)t * H_D + j] = f2bf(h);  // owner: 16 j's per WG
    }
    __syncthreads();
  }
}

// ------------------------------------------------------------- log-softmax
__global__ __launch_bounds__(256) void log_softmax64(const float* __restrict__ L,
                                                     float* __restrict__ out) {
  const int lane = threadIdx.x & 31;
  const int row = blockIdx.x * 8 + (threadIdx.x >> 5);
  const float* x = L + (long)row * T_D;
  float a = x[lane], b = x[lane + 32];
  float m = fmaxf(a, b);
#pragma unroll
  for (int off = 16; off > 0; off >>= 1) m = fmaxf(m, __shfl_xor(m, off, 32));
  float s = __expf(a - m) + __expf(b - m);
#pragma unroll
  for (int off = 16; off > 0; off >>= 1) s += __shfl_xor(s, off, 32);
  float lse = m + logf(s);
  out[(long)row * T_D + lane] = a - lse;
  out[(long)row * T_D + lane + 32] = b - lse;
}

// ---------------------------------------------------------------------------
extern "C" void kernel_launch(void* const* d_in, const int* in_sizes, int n_in,
                              void* d_out, int out_size, void* d_ws, size_t ws_size,
                              hipStream_t stream) {
  const int*   char_idx  = (const int*)  d_in[0];
  const int*   word_idx  = (const int*)  d_in[1];
  const float* W_emb_w   = (const float*)d_in[2];
  const float* W_emb_c   = (const float*)d_in[3];
  const float* W_ih_c    = (const float*)d_in[4];
  const float* W_hh_c    = (const float*)d_in[5];
  const float* b_ih_c    = (const float*)d_in[6];
  const float* b_hh_c    = (const float*)d_in[7];
  const float* W_ih      = (const float*)d_in[8];
  const float* W_hh      = (const float*)d_in[9];
  const float* b_ih      = (const float*)d_in[10];
  const float* b_hh      = (const float*)d_in[11];
  const float* W_tag     = (const float*)d_in[12];
  const float* b_tag     = (const float*)d_in[13];
  float* out = (float*)d_out;

  // ---- workspace layout (~179 MB total)
  char* ws = (char*)d_ws;
  size_t off = 0;
  auto alloc = [&](size_t bytes) -> void* {
    void* p = ws + off;
    off = (off + bytes + 255) & ~(size_t)255;
    return p;
  };
  unsigned short* wc_cat  = (unsigned short*)alloc((size_t)1024 * 512 * 2); // [4HC x (EC+HC)]
  float*          bias_c  = (float*)alloc(1024 * 4);
  unsigned short* wih_bf  = (unsigned short*)alloc((size_t)4096 * 1280 * 2);
  float*          bias_w  = (float*)alloc(4096 * 4);
  unsigned short* whh_bf  = (unsigned short*)alloc((size_t)4096 * 1024 * 2);
  unsigned short* wtag_bf = (unsigned short*)alloc((size_t)64 * 1024 * 2);
  unsigned short* A_char  = (unsigned short*)alloc((size_t)C_LEN * S_LEN * 512 * 2);
  float*          Gc      = (float*)alloc((size_t)S_LEN * 1024 * 4);
  float*          c_char  = (float*)alloc((size_t)S_LEN * HC_D * 4);
  unsigned short* A_w     = (unsigned short*)alloc((size_t)S_LEN * 1280 * 2);
  float*          P       = (float*)alloc((size_t)S_LEN * 4096 * 4);
  unsigned short* hs      = (unsigned short*)alloc((size_t)S_LEN * H_D * 2);
  float*          logits  = (float*)alloc((size_t)S_LEN * T_D * 4);
  float*          gatesb  = (float*)alloc((size_t)2 * 4096 * 4);   // ping-pong
  unsigned*       bar     = (unsigned*)alloc(256);
  (void)in_sizes; (void)n_in; (void)out_size; (void)ws_size;

  const int EB = 256;   // elementwise block
  const int EG = 2048;  // elementwise grid

  // ---- weight prep: bf16 conversions + [W_ih_c | W_hh_c] concat + bias sums
  cvt_bf16_strided<<<EG, EB, 0, stream>>>(W_ih_c, 256, wc_cat,       512, 1024, 256);
  cvt_bf16_strided<<<EG, EB, 0, stream>>>(W_hh_c, 256, wc_cat + 256, 512, 1024, 256);
  cvt_bf16_strided<<<EG, EB, 0, stream>>>(W_ih,  1280, wih_bf, 1280, 4096, 1280);
  cvt_bf16_strided<<<EG, EB, 0, stream>>>(W_hh,  1024, whh_bf, 1024, 4096, 1024);
  cvt_bf16_strided<<<EG, EB, 0, stream>>>(W_tag, 1024, wtag_bf, 1024, 64, 1024);
  add_bias_vec<<<8, EB, 0, stream>>>(b_ih_c, b_hh_c, bias_c, 1024);
  add_bias_vec<<<16, EB, 0, stream>>>(b_ih, b_hh, bias_w, 4096);

  // ---- embeddings
  embed_chars<<<EG, EB, 0, stream>>>(char_idx, W_emb_c, A_char);
  embed_words<<<EG, EB, 0, stream>>>(word_idx, W_emb_w, A_w);
  init_barrier<<<1, 32, 0, stream>>>(bar);

  // ---- char LSTM: 12 steps of [4096x512]x[512x1024] WMMA GEMM + cell update
  for (int t = 0; t < C_LEN; ++t) {
    const unsigned short* At = A_char + (size_t)t * S_LEN * 512;
    gemm_bf16_wmma<<<dim3(32, 16), 128, 0, stream>>>(
        At, 512, wc_cat, 512, bias_c, Gc, 1024, 512);
    unsigned short* Hdst = (t < C_LEN - 1)
        ? (A_char + (size_t)(t + 1) * S_LEN * 512 + 256)   // next step's h slot
        : (A_w + 1024);                                    // h_char into word input
    int ldH = (t < C_LEN - 1) ? 512 : 1280;
    lstm_cell_char<<<EG, EB, 0, stream>>>(Gc, c_char, Hdst, ldH, t == 0 ? 1 : 0);
  }

  // ---- word LSTM input precompute: P = [emb|h_char] @ W_ih^T, + (b_ih+b_hh) only on rows with row % 32 < 16
  gemm_bf16_wmma<<<dim3(32, 64), 128, 0, stream>>>(
      A_w, 1280, wih_bf, 1280, bias_w, P, 4096, 1280);

  // ---- sequential word recurrence (64 WGs, weights pinned in VGPRs)
  word_lstm_seq<<<NWG_SEQ, 256, 0, stream>>>(P, whh_bf, gatesb, hs, bar);

  // ---- tag projection + log-softmax
  gemm_bf16_wmma<<<dim3(32, 1), 128, 0, stream>>>(
      hs, 1024, wtag_bf, 1024, b_tag, logits, 64, 1024);
  log_softmax64<<<512, 256, 0, stream>>>(logits, out);
}
